// Model_49349174231298
// MI455X (gfx1250) — matrix operand order repaired and checked by reading, not compiled, not run on hardware
//
#include <hip/hip_runtime.h>
#include <hip/hip_bf16.h>

typedef __attribute__((ext_vector_type(16))) _Float16 v16h;
typedef __attribute__((ext_vector_type(8)))  _Float16 v8h;
typedef __attribute__((ext_vector_type(4)))  _Float16 v4h;
typedef __attribute__((ext_vector_type(8)))  float    v8f;

// ---------- helpers ----------
__device__ __forceinline__ float gelu_f(float x) {
    const float k0 = 0.7978845608028654f, k1 = 0.044715f;
    return 0.5f * x * (1.f + tanhf(k0 * (x + k1 * x * x * x)));
}
__device__ __forceinline__ unsigned encf(float f) {
    unsigned u = __float_as_uint(f);
    return (u & 0x80000000u) ? ~u : (u | 0x80000000u);
}
__device__ __forceinline__ float decf(unsigned u) {
    unsigned v = (u & 0x80000000u) ? (u & 0x7FFFFFFFu) : ~u;
    return __uint_as_float(v);
}
__device__ __forceinline__ v16h combine8(v8h lo, v8h hi) {
    return __builtin_shufflevector(lo, hi, 0,1,2,3,4,5,6,7,8,9,10,11,12,13,14,15);
}

// ---------- weight convert+transpose: W[mat][K][N] f32 -> Wt[mat][N][K] f16 ----------
// grid: (ceil(N/128), K, batch) -- no integer division
__global__ void convert_wt(const float* __restrict__ W, _Float16* __restrict__ Wt, int K, int N)
{
    int n = blockIdx.x * blockDim.x + threadIdx.x;
    if (n >= N) return;
    int k = blockIdx.y;
    long base = (long)blockIdx.z * K * N;
    Wt[base + (long)n * K + k] = (_Float16)W[base + (long)k * N + n];
}

// ---------- WMMA GEMM: Y = blend(act_out(act_in(X) @ W + b)) ----------
// X: nrows x 128 f32. Wt: f16 [ncols][128]. One 16-row tile per 32-thread block.
template<int NCT, int ACTIN, int ACTOUT, bool BLEND>
__global__ __launch_bounds__(32) void gemm16(
    const float* __restrict__ X, const _Float16* __restrict__ Wt,
    const float* __restrict__ bias, const float* __restrict__ xold,
    const float* __restrict__ skip_ptr, float* __restrict__ Y, int nrows)
{
    constexpr int ncols = NCT * 16;
    __shared__ __align__(16) _Float16 Xs[16 * 128];
    const int lane = threadIdx.x;
    const int half = lane >> 4, l15 = lane & 15;
    const int rb = blockIdx.x * 16;
    const bool full = (rb + 16 <= nrows);

    for (int i = lane; i < 16 * 32; i += 32) {
        int r = i >> 5, c4 = (i & 31) * 4;
        int gr = rb + r;
        if (!full && gr >= nrows) gr = nrows - 1;
        float4 v = *(const float4*)&X[(size_t)gr * 128 + c4];
        if (ACTIN == 1) { v.x = gelu_f(v.x); v.y = gelu_f(v.y); v.z = gelu_f(v.z); v.w = gelu_f(v.w); }
        v4h h4 = {(_Float16)v.x, (_Float16)v.y, (_Float16)v.z, (_Float16)v.w};
        *(v4h*)&Xs[r * 128 + c4] = h4;
    }
    __syncthreads();

    v16h afr[4];
#pragma unroll
    for (int kk = 0; kk < 4; ++kk) {
        const int ab = kk * 32 + (half ? 16 : 0);
        v8h lo = *(const v8h*)&Xs[l15 * 128 + ab];
        v8h hi = *(const v8h*)&Xs[l15 * 128 + ab + 8];
        afr[kk] = combine8(lo, hi);
    }

    float sa = 1.f, sb = 0.f;
    if (BLEND) { float s = 1.f / (1.f + __expf(-skip_ptr[0])); sa = s; sb = 1.f - s; }

#pragma unroll
    for (int ct = 0; ct < NCT; ++ct) {
        const int col = ct * 16 + l15;
        const _Float16* wrow = Wt + (size_t)col * 128;
        v16h bf[4];
#pragma unroll
        for (int kk = 0; kk < 4; ++kk)
            bf[kk] = *(const v16h*)&wrow[kk * 32 + (half ? 16 : 0)];
        v8f acc = {};
#pragma unroll
        for (int kk = 0; kk < 4; ++kk)
            acc = __builtin_amdgcn_wmma_f32_16x16x32_f16(false, afr[kk], false, bf[kk],
                                                         (short)0, acc, false, false);
        const float bv = bias[col];
        if (full) {
#pragma unroll
            for (int r = 0; r < 8; ++r) {
                int gr = rb + r + half * 8;
                float y = acc[r] + bv;
                if (ACTOUT == 1) y = fmaxf(y, 0.f);
                if (BLEND) y = sa * y + sb * xold[(size_t)gr * 128 + col];
                Y[(size_t)gr * ncols + col] = y;
            }
        } else {
#pragma unroll
            for (int r = 0; r < 8; ++r) {
                int gr = rb + r + half * 8;
                if (gr < nrows) {
                    float y = acc[r] + bv;
                    if (ACTOUT == 1) y = fmaxf(y, 0.f);
                    if (BLEND) y = sa * y + sb * xold[(size_t)gr * 128 + col];
                    Y[(size_t)gr * ncols + col] = y;
                }
            }
        }
    }
}

// ---------- edge logits: k_r = (k[src] 16x64) @ krel[h]; logit = k_r . q[dst] ----------
// krelT: f16, layout [h][e][d], 2*64*64 per (l,r)
__global__ __launch_bounds__(64) void edge_logits(
    const float* __restrict__ kqv_src, const float* __restrict__ kqv_dst,
    const int* __restrict__ src, const int* __restrict__ dst, int nE,
    const _Float16* __restrict__ krelT, const float* __restrict__ prel,
    float* __restrict__ lgt, unsigned* __restrict__ menc, int doff)
{
    __shared__ __align__(16) _Float16 Ksh[16 * 128];
    __shared__ float Rsh[2 * 16 * 64];
    const int tid = threadIdx.x;
    const int eb = blockIdx.x * 16;
    const bool full = (eb + 16 <= nE);

    for (int i = tid; i < 16 * 32; i += 64) {
        int e = i >> 5, c4 = (i & 31) * 4;
        int eg = eb + e;
        if (!full && eg >= nE) eg = nE - 1;
        float4 v = *(const float4*)&kqv_src[(size_t)src[eg] * 384 + c4];
        v4h h4 = {(_Float16)v.x, (_Float16)v.y, (_Float16)v.z, (_Float16)v.w};
        *(v4h*)&Ksh[e * 128 + c4] = h4;
    }
    __syncthreads();

    const int h = tid >> 5, lane = tid & 31;
    const int half = lane >> 4, l15 = lane & 15;

    v16h afr[2];
#pragma unroll
    for (int kk = 0; kk < 2; ++kk) {
        const int ab = kk * 32 + (half ? 16 : 0);
        v8h lo = *(const v8h*)&Ksh[l15 * 128 + h * 64 + ab];
        v8h hi = *(const v8h*)&Ksh[l15 * 128 + h * 64 + ab + 8];
        afr[kk] = combine8(lo, hi);
    }
#pragma unroll
    for (int ct = 0; ct < 4; ++ct) {
        const _Float16* wrow = krelT + (size_t)(h * 64 + ct * 16 + l15) * 64;
        v16h bf[2];
#pragma unroll
        for (int kk = 0; kk < 2; ++kk)
            bf[kk] = *(const v16h*)&wrow[kk * 32 + (half ? 16 : 0)];
        v8f acc = {};
#pragma unroll
        for (int kk = 0; kk < 2; ++kk)
            acc = __builtin_amdgcn_wmma_f32_16x16x32_f16(false, afr[kk], false, bf[kk],
                                                         (short)0, acc, false, false);
#pragma unroll
        for (int r = 0; r < 8; ++r)
            Rsh[h * 1024 + (r + half * 8) * 64 + ct * 16 + l15] = acc[r];
    }
    __syncthreads();

    if (lane < 16) {
        int eg = eb + lane;
        if (full || eg < nE) {
            int dn = dst[eg];
            const float4* q4 = (const float4*)&kqv_dst[(size_t)dn * 384 + 128 + h * 64];
            const float4* r4 = (const float4*)&Rsh[h * 1024 + lane * 64];
            float s = 0.f;
#pragma unroll
            for (int c = 0; c < 16; ++c) {
                float4 qv = q4[c], rv = r4[c];
                s += rv.x * qv.x + rv.y * qv.y + rv.z * qv.z + rv.w * qv.w;
            }
            float lg = s * 0.125f * prel[h];
            lgt[(size_t)eg * 2 + h] = lg;
            atomicMax(&menc[(size_t)(doff + dn) * 2 + h], encf(lg));
        }
    }
}

// ---------- exp(logit - m), accumulate denominator ----------
__global__ void exp_kernel(float* __restrict__ lg, const int* __restrict__ dst, int nE,
                           const unsigned* __restrict__ menc, float* __restrict__ den, int doff)
{
    int e = blockIdx.x * blockDim.x + threadIdx.x;
    if (e >= nE) return;
    int dn = doff + dst[e];
#pragma unroll
    for (int h = 0; h < 2; ++h) {
        float mm = decf(menc[(size_t)dn * 2 + h]);
        float ex = __expf(lg[(size_t)e * 2 + h] - mm);
        lg[(size_t)e * 2 + h] = ex;
        atomicAdd(&den[(size_t)dn * 2 + h], ex);
    }
}

// ---------- v_r = (v[src] 16x64)@(64x64); agg[dst] += alpha * v_r ----------
__global__ __launch_bounds__(64) void edge_agg(
    const float* __restrict__ kqv_src,
    const int* __restrict__ src, const int* __restrict__ dst, int nE,
    const _Float16* __restrict__ vrelT, const float* __restrict__ eexp,
    const float* __restrict__ den, float* __restrict__ agg, int doff)
{
    __shared__ __align__(16) _Float16 Vsh[16 * 128];
    __shared__ float Alph[2 * 16];
    __shared__ int   Dsh[16];
    const int tid = threadIdx.x;
    const int eb = blockIdx.x * 16;
    const bool full = (eb + 16 <= nE);

    for (int i = tid; i < 16 * 32; i += 64) {
        int e = i >> 5, c4 = (i & 31) * 4;
        int eg = eb + e;
        if (!full && eg >= nE) eg = nE - 1;
        float4 v = *(const float4*)&kqv_src[(size_t)src[eg] * 384 + 256 + c4];
        v4h h4 = {(_Float16)v.x, (_Float16)v.y, (_Float16)v.z, (_Float16)v.w};
        *(v4h*)&Vsh[e * 128 + c4] = h4;
    }
    if (tid < 32) {
        int h2 = tid >> 4, e = tid & 15;
        int eg = eb + e;
        if (!full && eg >= nE) eg = nE - 1;
        int dn = doff + dst[eg];
        if (h2 == 0) Dsh[e] = dn;
        float alpha = 0.f;
        if (full || eb + e < nE) {
            float dd = den[(size_t)dn * 2 + h2];
            alpha = (dd > 0.f) ? eexp[(size_t)(eb + e) * 2 + h2] / dd : 0.f;
        }
        Alph[h2 * 16 + e] = alpha;
    }
    __syncthreads();

    const int h = tid >> 5, lane = tid & 31;
    const int half = lane >> 4, l15 = lane & 15;

    v16h afr[2];
#pragma unroll
    for (int kk = 0; kk < 2; ++kk) {
        const int ab = kk * 32 + (half ? 16 : 0);
        v8h lo = *(const v8h*)&Vsh[l15 * 128 + h * 64 + ab];
        v8h hi = *(const v8h*)&Vsh[l15 * 128 + h * 64 + ab + 8];
        afr[kk] = combine8(lo, hi);
    }
#pragma unroll
    for (int ct = 0; ct < 4; ++ct) {
        const _Float16* wrow = vrelT + (size_t)(h * 64 + ct * 16 + l15) * 64;
        v16h bf[2];
#pragma unroll
        for (int kk = 0; kk < 2; ++kk)
            bf[kk] = *(const v16h*)&wrow[kk * 32 + (half ? 16 : 0)];
        v8f acc = {};
#pragma unroll
        for (int kk = 0; kk < 2; ++kk)
            acc = __builtin_amdgcn_wmma_f32_16x16x32_f16(false, afr[kk], false, bf[kk],
                                                         (short)0, acc, false, false);
        const int col = h * 64 + ct * 16 + l15;
        if (full) {
#pragma unroll
            for (int r = 0; r < 8; ++r) {
                int m = r + half * 8;
                atomicAdd(&agg[(size_t)Dsh[m] * 128 + col], Alph[h * 16 + m] * acc[r]);
            }
        } else {
#pragma unroll
            for (int r = 0; r < 8; ++r) {
                int m = r + half * 8;
                if (eb + m < nE)
                    atomicAdd(&agg[(size_t)Dsh[m] * 128 + col], Alph[h * 16 + m] * acc[r]);
            }
        }
    }
}

// ---------- final: out[i] = dot(xp[eli0[i]], xp[eli1[i]]) ----------
__global__ void dot_kernel(const float* __restrict__ xp, const int* __restrict__ eli,
                           int n, float* __restrict__ out)
{
    int g = blockIdx.x * blockDim.x + threadIdx.x;
    int pair = g >> 5;
    int lane = g & 31;
    if (pair >= n) return;
    int i0 = eli[pair];
    int i1 = eli[n + pair];
    float4 a = ((const float4*)&xp[(size_t)i0 * 128])[lane];
    float4 b = ((const float4*)&xp[(size_t)i1 * 128])[lane];
    float s = a.x * b.x + a.y * b.y + a.z * b.z + a.w * b.w;
#pragma unroll
    for (int off = 16; off > 0; off >>= 1)
        s += __shfl_xor(s, off, 32);
    if (lane == 0) out[pair] = s;
}

// ---------- host ----------
extern "C" void kernel_launch(void* const* d_in, const int* in_sizes, int n_in,
                              void* d_out, int out_size, void* d_ws, size_t ws_size,
                              hipStream_t stream)
{
    static const int NN[3]  = {300000, 200000, 5000};
    static const int OFF[4] = {0, 300000, 500000, 505000};
    static const int EST[6] = {1, 0, 0, 0, 0, 2};
    static const int EDT[6] = {0, 1, 0, 0, 2, 0};
    const int NT = 505000;

    const float* emb[3] = {(const float*)d_in[16], (const float*)d_in[17], (const float*)d_in[18]};
    const int* src[6]; const int* dst[6]; int ecnt[6];
    for (int r = 0; r < 6; ++r) {
        src[r] = (const int*)d_in[3 + 2 * r];
        dst[r] = (const int*)d_in[4 + 2 * r];
        ecnt[r] = in_sizes[3 + 2 * r];
    }
    const int* eli = (const int*)d_in[15];
    const int nlabel = in_sizes[15] / 2;
    const float* lin_w  = (const float*)d_in[19];
    const float* lin_b  = (const float*)d_in[20];
    const float* kqv_w  = (const float*)d_in[21];
    const float* kqv_b  = (const float*)d_in[22];
    const float* krel_w = (const float*)d_in[23];
    const float* vrel_w = (const float*)d_in[24];
    const float* prel   = (const float*)d_in[25];
    const float* out_w  = (const float*)d_in[26];
    const float* out_b  = (const float*)d_in[27];
    const float* skipp  = (const float*)d_in[28];

    float* w = (float*)d_ws;
    size_t o = 0;
    float* xsA = w + o;  o += (size_t)NT * 128;
    float* xsB = w + o;  o += (size_t)NT * 128;
    float* kqv = w + o;  o += (size_t)NT * 384;
    float* agg = w + o;  o += (size_t)NT * 128;
    unsigned* menc = (unsigned*)(w + o); o += (size_t)NT * 2;
    float* den = w + o;  o += (size_t)NT * 2;
    size_t eoff[7]; eoff[0] = 0;
    for (int r = 0; r < 6; ++r) eoff[r + 1] = eoff[r] + (size_t)ecnt[r];
    float* lgb = w + o;  o += eoff[6] * 2;

    // f16 transposed weights
    _Float16* hb = (_Float16*)(w + o);
    size_t ho = 0;
    _Float16* lin_wt = hb + ho; ho += (size_t)3 * 128 * 128;
    _Float16* kqv_wt = hb + ho; ho += (size_t)6 * 128 * 384;
    _Float16* out_wt = hb + ho; ho += (size_t)6 * 128 * 128;
    _Float16* krel_t = hb + ho; ho += (size_t)24 * 64 * 64;
    _Float16* vrel_t = hb + ho; ho += (size_t)24 * 64 * 64;

    // convert weights (tiny, L2-resident afterwards)
    convert_wt<<<dim3(1, 128, 3),  128, 0, stream>>>(lin_w,  lin_wt, 128, 128);
    convert_wt<<<dim3(3, 128, 6),  128, 0, stream>>>(kqv_w,  kqv_wt, 128, 384);
    convert_wt<<<dim3(1, 128, 6),  128, 0, stream>>>(out_w,  out_wt, 128, 128);
    convert_wt<<<dim3(1, 64, 24),  128, 0, stream>>>(krel_w, krel_t, 64, 64);
    convert_wt<<<dim3(1, 64, 24),  128, 0, stream>>>(vrel_w, vrel_t, 64, 64);

    // initial per-type embed -> relu(x @ lin_w + lin_b)
    for (int t = 0; t < 3; ++t) {
        gemm16<8, 0, 1, false><<<dim3((NN[t] + 15) / 16), 32, 0, stream>>>(
            emb[t], lin_wt + (size_t)t * 128 * 128, lin_b + t * 128,
            nullptr, nullptr, xsA + (size_t)OFF[t] * 128, NN[t]);
    }

    float* cur = xsA; float* nxt = xsB;
    for (int l = 0; l < 2; ++l) {
        for (int t = 0; t < 3; ++t) {
            gemm16<24, 0, 0, false><<<dim3((NN[t] + 15) / 16), 32, 0, stream>>>(
                cur + (size_t)OFF[t] * 128,
                kqv_wt + (size_t)(l * 3 + t) * 128 * 384, kqv_b + (size_t)(l * 3 + t) * 384,
                nullptr, nullptr, kqv + (size_t)OFF[t] * 384, NN[t]);
        }
        hipMemsetAsync(menc, 0, (size_t)NT * 2 * sizeof(unsigned), stream);
        hipMemsetAsync(den,  0, (size_t)NT * 2 * sizeof(float), stream);
        hipMemsetAsync(agg,  0, (size_t)NT * 128 * sizeof(float), stream);
        for (int r = 0; r < 6; ++r) {
            edge_logits<<<dim3((ecnt[r] + 15) / 16), 64, 0, stream>>>(
                kqv + (size_t)OFF[EST[r]] * 384, kqv + (size_t)OFF[EDT[r]] * 384,
                src[r], dst[r], ecnt[r],
                krel_t + (size_t)(l * 12 + r * 2) * 64 * 64, prel + (size_t)(l * 6 + r) * 2,
                lgb + eoff[r] * 2, menc, OFF[EDT[r]]);
        }
        for (int r = 0; r < 6; ++r) {
            exp_kernel<<<dim3((ecnt[r] + 255) / 256), 256, 0, stream>>>(
                lgb + eoff[r] * 2, dst[r], ecnt[r], menc, den, OFF[EDT[r]]);
        }
        for (int r = 0; r < 6; ++r) {
            edge_agg<<<dim3((ecnt[r] + 15) / 16), 64, 0, stream>>>(
                kqv + (size_t)OFF[EST[r]] * 384, src[r], dst[r], ecnt[r],
                vrel_t + (size_t)(l * 12 + r * 2) * 64 * 64,
                lgb + eoff[r] * 2, den, agg, OFF[EDT[r]]);
        }
        for (int t = 0; t < 3; ++t) {
            gemm16<8, 1, 0, true><<<dim3((NN[t] + 15) / 16), 32, 0, stream>>>(
                agg + (size_t)OFF[t] * 128,
                out_wt + (size_t)(l * 3 + t) * 128 * 128, out_b + (size_t)(l * 3 + t) * 128,
                cur + (size_t)OFF[t] * 128, skipp + (l * 3 + t),
                nxt + (size_t)OFF[t] * 128, NN[t]);
        }
        float* tmp = cur; cur = nxt; nxt = tmp;
    }

    long long nth = (long long)nlabel * 32;
    dot_kernel<<<dim3((unsigned)((nth + 255) / 256)), 256, 0, stream>>>(cur, eli, nlabel, (float*)d_out);

    (void)n_in; (void)out_size; (void)ws_size;
}
